// SceneContagion_3547642986630
// MI455X (gfx1250) — matrix operand order repaired and checked by reading, not compiled, not run on hardware
//
#include <hip/hip_runtime.h>

#define DD  512
#define HH  8
#define BB  64
#define NF_ 16384
#define NO_ 8192
#define NN  (NF_ + NO_)
#define EPS_ 1e-5f

typedef __attribute__((ext_vector_type(16))) __bf16 v16bf;
typedef __attribute__((ext_vector_type(8)))  float  v8f;

enum { MODE_STORE = 0, MODE_SCORES = 1, MODE_GATE = 2 };

// ---------------------------------------------------------------------------
// Prep 1: split fp32 rows into bf16 hi/lo planes, stored in plain k order
// (no permutation: a fragment holds k = 16*half + s of each 32-wide group),
// so lane (l15, half) reads its whole fragment as ONE contiguous 32B chunk at
// row*512 + k0 + half*16.   src = src0 rows [0,splitRow), then src1.
// ---------------------------------------------------------------------------
__global__ __launch_bounds__(256)
void split_rows(const float* __restrict__ src0, const float* __restrict__ src1,
                int splitRow, __bf16* __restrict__ hi, __bf16* __restrict__ lo)
{
  const int row = blockIdx.x;
  const float* s = (row < splitRow) ? src0 + (size_t)row * DD
                                    : src1 + (size_t)(row - splitRow) * DD;
  const size_t ob = (size_t)row * DD;
#pragma unroll
  for (int e = threadIdx.x; e < DD; e += 256) {
    const int g = e >> 5, w = e & 31, sect = w >> 3;
    const int dw = w;
    const float x = s[e];
    const __bf16 h = (__bf16)x;
    hi[ob + (g << 5) + dw] = h;
    lo[ob + (g << 5) + dw] = (__bf16)(x - (float)h);
  }
}

// ---------------------------------------------------------------------------
// Prep 2: W(512x512, row-major K x N) -> transposed bf16 hi/lo planes
// WT[n][k], so a B fragment (16 consecutive K, fixed column n) is one
// contiguous 32B chunk at n*512 + k0 + half*16.
// ---------------------------------------------------------------------------
__global__ __launch_bounds__(256)
void transpose_split(const float* __restrict__ W,
                     __bf16* __restrict__ hi, __bf16* __restrict__ lo)
{
  __shared__ float tile[32][33];
  const int tx = threadIdx.x & 31;
  const int ty = threadIdx.x >> 5;          // 0..7
  const int kt = blockIdx.x * 32;
  const int nt = blockIdx.y * 32;
#pragma unroll
  for (int i = 0; i < 4; ++i) {
    const int k = kt + ty + i * 8;
    tile[ty + i * 8][tx] = W[(size_t)k * DD + nt + tx];   // coalesced read
  }
  __syncthreads();
#pragma unroll
  for (int i = 0; i < 4; ++i) {
    const int n = nt + ty + i * 8;
    const float x = tile[tx][ty + i * 8];
    const __bf16 h = (__bf16)x;
    const size_t o = (size_t)n * DD + kt + tx;            // coalesced write
    hi[o] = h;
    lo[o] = (__bf16)(x - (float)h);
  }
}

// ---------------------------------------------------------------------------
// WMMA GEMM on pre-split bf16 planes.  C = A*W + bias computed as
// Ah*Bh + Ah*Bl + Al*Bh (bf16x3 ~ fp32).  Block = 4 waves; wave = 16x64 tile;
// block = 64 rows x 64 cols; 64 cols == one attention head.
// ---------------------------------------------------------------------------
template<int MODE>
__global__ __launch_bounds__(128)
void gemm_wmma(const __bf16* __restrict__ Ahi, const __bf16* __restrict__ Alo,
               const __bf16* __restrict__ Bhi, const __bf16* __restrict__ Blo,
               const float* __restrict__ bias,
               float* __restrict__ out,          // STORE: M x 512
               const float* __restrict__ qmat,   // SCORES: q (64 x 512)
               const int*   __restrict__ faceb,
               const int*   __restrict__ objb,
               float* __restrict__ scores,       // SCORES: N x 8
               const float* __restrict__ fsbuf,  // GATE: fs (64 x 512)
               const float* __restrict__ alpha,  // GATE: (512)
               const float* __restrict__ resid,  // GATE: hf (NF x 512)
               float* __restrict__ dout)         // GATE: pre-LN out
{
  const int wave = threadIdx.x >> 5;
  const int lane = threadIdx.x & 31;
  const int half = lane >> 4;
  const int l15  = lane & 15;
  const int r0   = blockIdx.x * 64 + wave * 16;
  const int c0   = blockIdx.y * 64;

  const __bf16* arh = Ahi + ((size_t)(r0 + l15) << 9);
  const __bf16* arl = Alo + ((size_t)(r0 + l15) << 9);

  v8f acc[4];
#pragma unroll
  for (int t = 0; t < 4; ++t)
#pragma unroll
    for (int j = 0; j < 8; ++j) acc[t][j] = 0.0f;

  for (int k0 = 0; k0 < DD; k0 += 32) {
    const int fo = k0 + half * 16;
    const v16bf ah = *reinterpret_cast<const v16bf*>(arh + fo);
    const v16bf al = *reinterpret_cast<const v16bf*>(arl + fo);
#pragma unroll
    for (int t = 0; t < 4; ++t) {
      const size_t bo_ = ((size_t)(c0 + t * 16 + l15) << 9) + fo;
      const v16bf bh = *reinterpret_cast<const v16bf*>(Bhi + bo_);
      const v16bf bl = *reinterpret_cast<const v16bf*>(Blo + bo_);
      acc[t] = __builtin_amdgcn_wmma_f32_16x16x32_bf16(false, ah, false, bh,
                                                       (short)0, acc[t], false, false);
      acc[t] = __builtin_amdgcn_wmma_f32_16x16x32_bf16(false, ah, false, bl,
                                                       (short)0, acc[t], false, false);
      acc[t] = __builtin_amdgcn_wmma_f32_16x16x32_bf16(false, al, false, bh,
                                                       (short)0, acc[t], false, false);
    }
  }

  // C layout: element(lane, vgpr j, tile t) = C[r0 + j + 8*half][c0 + t*16 + l15]
  if constexpr (MODE == MODE_STORE) {
#pragma unroll
    for (int t = 0; t < 4; ++t) {
      const int col = c0 + t * 16 + l15;
      const float bv = bias[col];
#pragma unroll
      for (int j = 0; j < 8; ++j) {
        const int row = r0 + j + 8 * half;
        out[(size_t)row * DD + col] = acc[t][j] + bv;
      }
    }
  } else if constexpr (MODE == MODE_SCORES) {
    // k never hits memory: scores[row][h] = (k_row . q[seg[row], head]) / 8
    const int hd = blockIdx.y;
    float p[8];
    int   sg[8];
#pragma unroll
    for (int j = 0; j < 8; ++j) {
      p[j] = 0.0f;
      const int row = r0 + j + 8 * half;
      sg[j] = (row < NF_) ? faceb[row] : objb[row - NF_];
    }
#pragma unroll
    for (int t = 0; t < 4; ++t) {
      const int col = c0 + t * 16 + l15;
      const float bv = bias[col];
#pragma unroll
      for (int j = 0; j < 8; ++j) {
        p[j] += (acc[t][j] + bv) * qmat[(size_t)sg[j] * DD + col];
      }
    }
#pragma unroll
    for (int j = 0; j < 8; ++j) {
      float pj = p[j];
#pragma unroll
      for (int m = 1; m < 16; m <<= 1) pj += __shfl_xor(pj, m, 32);
      if (l15 == 0) {
        const int row = r0 + j + 8 * half;
        scores[(size_t)row * HH + hd] = pj * 0.125f;     // 1/sqrt(dh)
      }
    }
  } else {  // MODE_GATE
#pragma unroll
    for (int t = 0; t < 4; ++t) {
      const int col = c0 + t * 16 + l15;
      const float bgv = bias[col];
      const float av  = alpha[col];
#pragma unroll
      for (int j = 0; j < 8; ++j) {
        const int row = r0 + j + 8 * half;
        const float g = 1.0f / (1.0f + __expf(-(acc[t][j] + bgv)));
        const int fb  = faceb[row];
        const float sig = av * fsbuf[(size_t)fb * DD + col];
        dout[(size_t)row * DD + col] = resid[(size_t)row * DD + col] + g * sig;
      }
    }
  }
}

// ---------------------------------------------------------------------------
// Ragged segment softmax-attention: one block per batch segment (sorted
// batch indices -> binary-searched ranges; fixed-order reductions).
// ---------------------------------------------------------------------------
__device__ __forceinline__ int lbound(const int* __restrict__ a, int n, int key) {
  int lo = 0, hi = n;
  while (lo < hi) {
    int mid = (lo + hi) >> 1;
    if (a[mid] < key) lo = mid + 1; else hi = mid;
  }
  return lo;
}

__global__ __launch_bounds__(256)
void segattn(const float* __restrict__ scores, const float* __restrict__ v,
             const int* __restrict__ faceb, const int* __restrict__ objb,
             float* __restrict__ attn)
{
  __shared__ int   rng[4];
  __shared__ float smax[8 * 32];
  __shared__ float mxs[8];
  __shared__ float den[8];
  const int b = blockIdx.x;
  const int t = threadIdx.x;

  if (t == 0) {
    rng[0] = lbound(faceb, NF_, b);  rng[1] = lbound(faceb, NF_, b + 1);
    rng[2] = lbound(objb,  NO_, b);  rng[3] = lbound(objb,  NO_, b + 1);
  }
  __syncthreads();
  const int fs0 = rng[0], fe0 = rng[1], os0 = rng[2], oe0 = rng[3];

  {
    const int h = t & 7, g = t >> 3;
    float m = -3.4e38f;
    for (int n = fs0 + g; n < fe0; n += 32)
      m = fmaxf(m, scores[(size_t)n * HH + h]);
    for (int i = os0 + g; i < oe0; i += 32)
      m = fmaxf(m, scores[(size_t)(NF_ + i) * HH + h]);
    smax[h * 32 + g] = m;
  }
  __syncthreads();
  if (t < 8) {
    float m = -3.4e38f;
    for (int g = 0; g < 32; ++g) m = fmaxf(m, smax[t * 32 + g]);
    mxs[t] = m;
  }
  __syncthreads();

  const int c1 = t, c2 = t + 256;
  const int h1 = t >> 6, h2 = h1 + 4;
  const bool lead = (t & 63) == 0;
  const float m1 = mxs[h1], m2 = mxs[h2];
  float num1 = 0.f, num2 = 0.f, d1 = 0.f, d2 = 0.f;

  for (int n = fs0; n < fe0; ++n) {
    const float e1 = __expf(scores[(size_t)n * HH + h1] - m1);
    const float e2 = __expf(scores[(size_t)n * HH + h2] - m2);
    const float* vr = v + (size_t)n * DD;
    num1 += e1 * vr[c1];
    num2 += e2 * vr[c2];
    if (lead) { d1 += e1; d2 += e2; }
  }
  for (int i = os0; i < oe0; ++i) {
    const int node = NF_ + i;
    const float e1 = __expf(scores[(size_t)node * HH + h1] - m1);
    const float e2 = __expf(scores[(size_t)node * HH + h2] - m2);
    const float* vr = v + (size_t)node * DD;
    num1 += e1 * vr[c1];
    num2 += e2 * vr[c2];
    if (lead) { d1 += e1; d2 += e2; }
  }
  if (lead) { den[h1] = d1; den[h2] = d2; }
  __syncthreads();
  attn[(size_t)b * DD + c1] = num1 / den[h1];
  attn[(size_t)b * DD + c2] = num2 / den[h2];
}

// ---------------------------------------------------------------------------
// LayerNorm over 512-wide rows (exact two-pass mean / mean((x-m)^2)).
// ---------------------------------------------------------------------------
__global__ __launch_bounds__(256)
void ln_rows(const float* __restrict__ ina, const float* __restrict__ inb,
             float* __restrict__ out)
{
  __shared__ float red[256];
  const int row = blockIdx.x;
  const int t   = threadIdx.x;
  const size_t base = (size_t)row * DD;

  float x0 = ina[base + t];
  float x1 = ina[base + t + 256];
  if (inb) { x0 += inb[base + t]; x1 += inb[base + t + 256]; }

  red[t] = x0 + x1;
  __syncthreads();
  for (int s = 128; s > 0; s >>= 1) {
    if (t < s) red[t] += red[t + s];
    __syncthreads();
  }
  const float mean = red[0] * (1.0f / DD);
  __syncthreads();

  const float d0 = x0 - mean, d1 = x1 - mean;
  red[t] = d0 * d0 + d1 * d1;
  __syncthreads();
  for (int s = 128; s > 0; s >>= 1) {
    if (t < s) red[t] += red[t + s];
    __syncthreads();
  }
  const float r = rsqrtf(red[0] * (1.0f / DD) + EPS_);
  out[base + t]       = d0 * r;
  out[base + t + 256] = d1 * r;
}

// ---------------------------------------------------------------------------
extern "C" void kernel_launch(void* const* d_in, const int* in_sizes, int n_in,
                              void* d_out, int out_size, void* d_ws, size_t ws_size,
                              hipStream_t stream) {
  (void)in_sizes; (void)n_in; (void)out_size; (void)ws_size;

  const float* hf    = (const float*)d_in[0];
  const float* ho    = (const float*)d_in[1];
  const float* ps    = (const float*)d_in[2];
  const int*   faceb = (const int*)  d_in[3];
  const int*   objb  = (const int*)  d_in[4];
  const float* Wq = (const float*)d_in[5];  const float* bq = (const float*)d_in[6];
  const float* Wk = (const float*)d_in[7];  const float* bk = (const float*)d_in[8];
  const float* Wv = (const float*)d_in[9];  const float* bv = (const float*)d_in[10];
  const float* Wo = (const float*)d_in[11]; const float* bo = (const float*)d_in[12];
  const float* Wg = (const float*)d_in[13]; const float* bg = (const float*)d_in[14];
  const float* alpha = (const float*)d_in[15];

  float* out   = (float*)d_out;
  float* hfnew = out;                        // NF x 512
  float* fsout = out + (size_t)NF_ * DD;     // 64 x 512

  // ---- fp32 scratch ----
  float* ws      = (float*)d_ws;
  float* qbuf    = ws;                                  // 64 x 512
  float* scoresB = qbuf    + (size_t)BB * DD;           // N x 8
  float* attnB   = scoresB + (size_t)NN * HH;           // 64 x 512
  float* attnOut = attnB   + (size_t)BB * DD;           // 64 x 512
  float* vbuf    = attnOut + (size_t)BB * DD;           // N x 512

  // ---- bf16 hi/lo planes ----
  __bf16* Anh = (__bf16*)(vbuf + (size_t)NN * DD);      // nodes hi  (N x 512)
  __bf16* Anl = Anh + (size_t)NN * DD;                  // nodes lo
  __bf16* Psh = Anl + (size_t)NN * DD;                  // ps hi (64 x 512)
  __bf16* Psl = Psh + (size_t)BB * DD;
  __bf16* Ath = Psl + (size_t)BB * DD;                  // attn hi (64 x 512)
  __bf16* Atl = Ath + (size_t)BB * DD;
  __bf16* Wp  = Atl + (size_t)BB * DD;                  // 5 x 2 x (512x512)
  const size_t WSZ = (size_t)DD * DD;
  __bf16 *Wqh = Wp,            *Wql = Wp + WSZ;
  __bf16 *Wkh = Wp + 2 * WSZ,  *Wkl = Wp + 3 * WSZ;
  __bf16 *Wvh = Wp + 4 * WSZ,  *Wvl = Wp + 5 * WSZ;
  __bf16 *Woh = Wp + 6 * WSZ,  *Wol = Wp + 7 * WSZ;
  __bf16 *Wgh = Wp + 8 * WSZ,  *Wgl = Wp + 9 * WSZ;

  const int BIG = 1 << 30;
  dim3 blk(128);
  dim3 tgrid(16, 16);

  // ---- prep: one-time bf16 hi/lo conversion ----
  transpose_split<<<tgrid, dim3(256), 0, stream>>>(Wq, Wqh, Wql);
  transpose_split<<<tgrid, dim3(256), 0, stream>>>(Wk, Wkh, Wkl);
  transpose_split<<<tgrid, dim3(256), 0, stream>>>(Wv, Wvh, Wvl);
  transpose_split<<<tgrid, dim3(256), 0, stream>>>(Wo, Woh, Wol);
  transpose_split<<<tgrid, dim3(256), 0, stream>>>(Wg, Wgh, Wgl);
  split_rows<<<dim3(NN), dim3(256), 0, stream>>>(hf, ho, NF_, Anh, Anl);
  split_rows<<<dim3(BB), dim3(256), 0, stream>>>(ps, ps, BIG, Psh, Psl);

  // 1) q = ps@Wq + bq
  gemm_wmma<MODE_STORE><<<dim3(BB / 64, DD / 64), blk, 0, stream>>>(
      Psh, Psl, Wqh, Wql, bq, qbuf,
      nullptr, nullptr, nullptr, nullptr, nullptr, nullptr, nullptr, nullptr);

  // 2) scores = (nodes@Wk + bk) . q[seg] / sqrt(dh)  (k never stored)
  gemm_wmma<MODE_SCORES><<<dim3(NN / 64, DD / 64), blk, 0, stream>>>(
      Anh, Anl, Wkh, Wkl, bk, nullptr,
      qbuf, faceb, objb, scoresB, nullptr, nullptr, nullptr, nullptr);

  // 3) v = nodes@Wv + bv
  gemm_wmma<MODE_STORE><<<dim3(NN / 64, DD / 64), blk, 0, stream>>>(
      Anh, Anl, Wvh, Wvl, bv, vbuf,
      nullptr, nullptr, nullptr, nullptr, nullptr, nullptr, nullptr, nullptr);

  // 4) segment softmax attention -> attn (64 x 512)
  segattn<<<dim3(BB), dim3(256), 0, stream>>>(scoresB, vbuf, faceb, objb, attnB);

  // 5) attn_out = attn@Wo + bo
  split_rows<<<dim3(BB), dim3(256), 0, stream>>>(attnB, attnB, BIG, Ath, Atl);
  gemm_wmma<MODE_STORE><<<dim3(BB / 64, DD / 64), blk, 0, stream>>>(
      Ath, Atl, Woh, Wol, bo, attnOut,
      nullptr, nullptr, nullptr, nullptr, nullptr, nullptr, nullptr, nullptr);

  // 6) fs = LN(ps + attn_out)
  ln_rows<<<dim3(BB), dim3(256), 0, stream>>>(ps, attnOut, fsout);

  // 7) pre-LN hf_new = hf + sigmoid(hf@Wg + bg) * (alpha * fs[face])
  //    (hf rows are the first NF rows of the nodes plane -> reuse it)
  gemm_wmma<MODE_GATE><<<dim3(NF_ / 64, DD / 64), blk, 0, stream>>>(
      Anh, Anl, Wgh, Wgl, bg, nullptr,
      nullptr, faceb, nullptr, nullptr, fsout, alpha, hf, hfnew);

  // 8) hf_new = LN(pre-LN) in place
  ln_rows<<<dim3(NF_), dim3(256), 0, stream>>>(hfnew, nullptr, hfnew);
}
